// MessageBlock_77343771066913
// MI455X (gfx1250) — compile-verified, matrix operand order checked by reading, not run on hardware
//
#include <hip/hip_runtime.h>
#include <hip/hip_bf16.h>

// Problem constants (match reference)
#define N_NODE 5000
#define SB     49
#define SBR    19
#define YROT   2
#define CCH    128
#define HID    256
#define NBF    512
#define EEDGE  10000
#define EY     (EEDGE * YROT)      // 20000
#define F2     (2 * SBR * CCH)     // 4864
#define F1     (SBR * CCH)         // 2432
#define SBP    64                  // SB padded to K granularity
#define SBRP   32                  // SBR padded
#define SBMP   64                  // SB padded as M dim (rotate_inv)

typedef __attribute__((ext_vector_type(16))) __bf16 v16bf;
typedef __attribute__((ext_vector_type(8)))  float  v8f;

static __device__ __forceinline__ float silu_f(float x) {
    return x / (1.0f + __expf(-x));
}

// ---------------------------------------------------------------------------
// Conversion kernels (one-time per launch). These pick bf16 layouts so every
// WMMA fragment load downstream is a contiguous, 32B-aligned vector load.
// ---------------------------------------------------------------------------

// plain f32 -> bf16
__global__ void cvt_f32_bf16(const float* __restrict__ src, __bf16* __restrict__ dst, long n) {
    long i = (long)blockIdx.x * blockDim.x + threadIdx.x;
    long stride = (long)gridDim.x * blockDim.x;
    for (; i < n; i += stride) dst[i] = (__bf16)src[i];
}

// weight [K,N] f32 -> [N,K] bf16 (transposed so B-fragments are contiguous)
__global__ void cvt_transpose(const float* __restrict__ src, __bf16* __restrict__ dst,
                              int K, int N) {
    long n = (long)K * N;
    long i = (long)blockIdx.x * blockDim.x + threadIdx.x;
    long stride = (long)gridDim.x * blockDim.x;
    for (; i < n; i += stride) {
        long col = i / K, k = i % K;             // dst[col][k]
        dst[i] = (__bf16)src[k * (long)N + col];
    }
}

// batched pad: dst[b][r<RP][c<CP] = (r<R && c<C) ? src[b][r][c] : 0
__global__ void cvt_pad(const float* __restrict__ src, __bf16* __restrict__ dst,
                        long B, int R, int C, int RP, int CP) {
    long n = B * RP * CP;
    long i = (long)blockIdx.x * blockDim.x + threadIdx.x;
    long stride = (long)gridDim.x * blockDim.x;
    for (; i < n; i += stride) {
        long b = i / ((long)RP * CP);
        int  rem = (int)(i - b * RP * CP);
        int  r = rem / CP, c = rem % CP;
        dst[i] = (r < R && c < C) ? (__bf16)src[(b * R + r) * C + c] : (__bf16)0.0f;
    }
}

// x [node][SB=49][C=128] f32 -> xT [node][C=128][SBP=64] bf16 (transpose + pad)
__global__ void cvt_x_tpad(const float* __restrict__ src, __bf16* __restrict__ dst) {
    long n = (long)N_NODE * CCH * SBP;
    long i = (long)blockIdx.x * blockDim.x + threadIdx.x;
    long stride = (long)gridDim.x * blockDim.x;
    for (; i < n; i += stride) {
        long b = i / (CCH * SBP);
        int  rem = (int)(i - b * CCH * SBP);
        int  c = rem / SBP, k = rem % SBP;
        dst[i] = (k < SB) ? (__bf16)src[(b * SB + k) * CCH + c] : (__bf16)0.0f;
    }
}

// ---------------------------------------------------------------------------
// Generic bf16 WMMA GEMM: C = epilogue(A[M,K] @ B[K,N] + bias[N])
//   A row-major [M,K]; BT row-major [N,K] (pre-transposed weights).
//   K % 64 == 0; N % 128 == 0; M guarded.
//   Block = 256 threads (8 waves), tile 128x128, K staged 64-deep (16 WMMA/stage).
//   mode 0: silu -> f32 out
//   mode 1: silu * gate[(m>>1)*N+n] -> bf16 out
//   mode 2: silu -> bf16 out
// ---------------------------------------------------------------------------
__global__ __launch_bounds__(256) void gemm_bf16_silu(
    const __bf16* __restrict__ A, const __bf16* __restrict__ BT,
    const float* __restrict__ bias, const float* __restrict__ gate,
    void* __restrict__ Cout, int M, int N, int K, int mode)
{
    __shared__ __bf16 As[128 * 64];    // [row][k], 16 KB
    __shared__ __bf16 Bs[128 * 64];    // [col][k], 16 KB

    const int tid  = threadIdx.x;
    const int lane = tid & 31;
    const int w    = tid >> 5;
    const int blockM = blockIdx.x * 128;
    const int blockN = blockIdx.y * 128;
    const int wm = (w >> 1) * 32;      // wave row offset in tile
    const int wn = (w & 1) * 64;       // wave col offset in tile
    const int lm = lane & 15;
    const int kh = (lane >> 4) * 16;   // 16-half K group select

    v8f acc[2][4] = {};

    for (int k0 = 0; k0 < K; k0 += 64) {
        // Stage A tile 128x64: 1024 uint4 loads, 4 per thread
        #pragma unroll
        for (int it = 0; it < 4; ++it) {
            int li  = it * 256 + tid;
            int row = li >> 3;
            int kc  = (li & 7) * 8;
            uint4 v = make_uint4(0u, 0u, 0u, 0u);
            int gr = blockM + row;
            if (gr < M) v = *(const uint4*)(A + (size_t)gr * K + k0 + kc);
            *(uint4*)(As + row * 64 + kc) = v;
        }
        // Stage BT tile 128x64: contiguous along K (pre-transposed), no guards
        #pragma unroll
        for (int it = 0; it < 4; ++it) {
            int li  = it * 256 + tid;
            int row = li >> 3;
            int kc  = (li & 7) * 8;
            uint4 v = *(const uint4*)(BT + (size_t)(blockN + row) * K + k0 + kc);
            *(uint4*)(Bs + row * 64 + kc) = v;
        }
        __syncthreads();

        if (k0 + 64 < K) {
            __builtin_prefetch(BT + (size_t)(blockN + tid / 2) * K + k0 + 64 + (tid & 1) * 32, 0, 1);
            __builtin_prefetch(A + (size_t)(blockM + tid / 2) * K + k0 + 64 + (tid & 1) * 32, 0, 1);
        }

        v16bf af[2][2];
        #pragma unroll
        for (int t = 0; t < 2; ++t)
            #pragma unroll
            for (int ks = 0; ks < 2; ++ks)
                af[t][ks] = *(const v16bf*)(As + (wm + t * 16 + lm) * 64 + ks * 32 + kh);

        #pragma unroll
        for (int u = 0; u < 4; ++u) {
            #pragma unroll
            for (int ks = 0; ks < 2; ++ks) {
                v16bf bfrag = *(const v16bf*)(Bs + (wn + u * 16 + lm) * 64 + ks * 32 + kh);
                acc[0][u] = __builtin_amdgcn_wmma_f32_16x16x32_bf16(
                    false, af[0][ks], false, bfrag, (short)0, acc[0][u], false, false);
                acc[1][u] = __builtin_amdgcn_wmma_f32_16x16x32_bf16(
                    false, af[1][ks], false, bfrag, (short)0, acc[1][u], false, false);
            }
        }
        __syncthreads();
    }

    // Epilogue: C/D frag layout -> VGPR i: lanes0-15 M=i, lanes16-31 M=i+8
    const int mo = (lane >> 4) * 8;
    #pragma unroll
    for (int t = 0; t < 2; ++t) {
        #pragma unroll
        for (int u = 0; u < 4; ++u) {
            #pragma unroll
            for (int i = 0; i < 8; ++i) {
                int m = blockM + wm + t * 16 + i + mo;
                int n = blockN + wn + u * 16 + lm;
                if (m < M) {
                    float v = acc[t][u][i] + bias[n];
                    v = silu_f(v);
                    if (mode == 1) v *= gate[(size_t)(m >> 1) * N + n];
                    if (mode == 0) ((float*)Cout)[(size_t)m * N + n] = v;
                    else           ((__bf16*)Cout)[(size_t)m * N + n] = (__bf16)v;
                }
            }
        }
    }
}

// ---------------------------------------------------------------------------
// Rotate: rot[ey][half*F1 + r*C + c] = sum_s wigner[ey][r][s] * x[node][s][c]
//   A = wig_pad[ey] (32x64, zero-padded), B^T = xT[node] (128x64, zero-padded)
//   All fragments are contiguous 32B vector loads. Block = 64 threads (2 waves).
// ---------------------------------------------------------------------------
__global__ __launch_bounds__(64) void rotate_wmma(
    const __bf16* __restrict__ wig_pad, const __bf16* __restrict__ xT,
    const int* __restrict__ edge_index, __bf16* __restrict__ rot)
{
    const int ey   = blockIdx.x;
    const int half = blockIdx.y;            // 0 = source, 1 = target
    const int e    = ey / YROT;
    const int node = edge_index[half * EEDGE + e];

    const __bf16* A = wig_pad + (size_t)ey * SBRP * SBP;        // [32][64]
    const __bf16* B = xT      + (size_t)node * CCH * SBP;       // [128][64]

    const int lane = threadIdx.x & 31;
    const int w    = threadIdx.x >> 5;      // M-tile (0,1)
    const int lm   = lane & 15;
    const int kg   = (lane >> 4) * 16;

    v16bf af[2];
    #pragma unroll
    for (int ks = 0; ks < 2; ++ks)
        af[ks] = *(const v16bf*)(A + (w * 16 + lm) * SBP + ks * 32 + kg);

    __bf16* out = rot + (size_t)ey * F2 + (size_t)half * F1;
    const int mo = (lane >> 4) * 8;

    #pragma unroll
    for (int nt = 0; nt < 8; ++nt) {
        v8f acc = {};
        #pragma unroll
        for (int ks = 0; ks < 2; ++ks) {
            v16bf bf = *(const v16bf*)(B + (nt * 16 + lm) * SBP + ks * 32 + kg);
            acc = __builtin_amdgcn_wmma_f32_16x16x32_bf16(
                false, af[ks], false, bf, (short)0, acc, false, false);
        }
        #pragma unroll
        for (int i = 0; i < 8; ++i) {
            int m = w * 16 + i + mo;
            if (m < SBR) out[m * CCH + nt * 16 + lm] = (__bf16)acc[i];
        }
    }
}

// ---------------------------------------------------------------------------
// Mean over Y=2 copies, written transposed+padded for the inverse rotation:
//   meanT[e][c][k<32] = (k<19) ? 0.5*(msg[2e][k*128+c] + msg[2e+1][k*128+c]) : 0
// ---------------------------------------------------------------------------
__global__ void mean_y_tpad(const __bf16* __restrict__ msg, __bf16* __restrict__ meanT) {
    long n = (long)EEDGE * CCH * SBRP;
    long i = (long)blockIdx.x * blockDim.x + threadIdx.x;
    long stride = (long)gridDim.x * blockDim.x;
    for (; i < n; i += stride) {
        long e = i / (CCH * SBRP);
        int  rem = (int)(i - e * CCH * SBRP);
        int  c = rem / SBRP, k = rem % SBRP;
        float v = 0.0f;
        if (k < SBR) {
            float a = (float)msg[(e * 2 + 0) * (long)F1 + k * CCH + c];
            float b = (float)msg[(e * 2 + 1) * (long)F1 + k * CCH + c];
            v = 0.5f * (a + b);
        }
        meanT[i] = (__bf16)v;
    }
}

// ---------------------------------------------------------------------------
// Inverse rotation: out[e] = wigner_inv[e] (49x19) @ mean[e] (19x128), f32 out
//   A = wiginv_pad[e] (64x32), B^T = meanT[e] (128x32). Block = 128 (4 waves).
// ---------------------------------------------------------------------------
__global__ __launch_bounds__(128) void rotate_inv_wmma(
    const __bf16* __restrict__ wiginv_pad, const __bf16* __restrict__ meanT,
    float* __restrict__ out)
{
    const int e = blockIdx.x;
    const __bf16* A = wiginv_pad + (size_t)e * SBMP * SBRP;   // [64][32]
    const __bf16* B = meanT      + (size_t)e * CCH * SBRP;    // [128][32]

    const int lane = threadIdx.x & 31;
    const int w    = threadIdx.x >> 5;   // M-tile 0..3
    const int lm   = lane & 15;
    const int kg   = (lane >> 4) * 16;

    v16bf af = *(const v16bf*)(A + (w * 16 + lm) * SBRP + kg);

    float* dst = out + (size_t)e * SB * CCH;
    const int mo = (lane >> 4) * 8;

    #pragma unroll
    for (int nt = 0; nt < 8; ++nt) {
        v16bf bf = *(const v16bf*)(B + (nt * 16 + lm) * SBRP + kg);
        v8f acc = {};
        acc = __builtin_amdgcn_wmma_f32_16x16x32_bf16(
            false, af, false, bf, (short)0, acc, false, false);
        #pragma unroll
        for (int i = 0; i < 8; ++i) {
            int m = w * 16 + i + mo;
            if (m < SB) dst[m * CCH + nt * 16 + lm] = acc[i];
        }
    }
}

// ---------------------------------------------------------------------------
// Host-side launcher
// ---------------------------------------------------------------------------
static inline size_t align256(size_t x) { return (x + 255) & ~(size_t)255; }

extern "C" void kernel_launch(void* const* d_in, const int* in_sizes, int n_in,
                              void* d_out, int out_size, void* d_ws, size_t ws_size,
                              hipStream_t stream) {
    const float* x        = (const float*)d_in[0];
    const float* x_edge   = (const float*)d_in[1];
    const int*   eidx     = (const int*)  d_in[2];
    const float* wigner   = (const float*)d_in[3];
    const float* wiginv   = (const float*)d_in[4];
    const float* w_dist   = (const float*)d_in[5];
    const float* b_dist   = (const float*)d_in[6];
    const float* w_proj   = (const float*)d_in[7];
    const float* b_proj   = (const float*)d_in[8];
    const float* w_edge   = (const float*)d_in[9];
    const float* b_edge   = (const float*)d_in[10];
    const float* w_out    = (const float*)d_in[11];
    const float* b_out    = (const float*)d_in[12];
    float* out = (float*)d_out;
    (void)in_sizes; (void)n_in; (void)out_size; (void)ws_size;

    // Workspace carve-up
    char* ws = (char*)d_ws;
    size_t off = 0;
    auto carve = [&](size_t bytes) { char* p = ws + off; off += align256(bytes); return p; };

    __bf16* xT        = (__bf16*)carve((size_t)N_NODE * CCH * SBP * 2);   // 82 MB
    __bf16* xe_bf     = (__bf16*)carve((size_t)EEDGE * NBF * 2);
    __bf16* wig_pad   = (__bf16*)carve((size_t)EY * SBRP * SBP * 2);      // 82 MB
    __bf16* wiginv_p  = (__bf16*)carve((size_t)EEDGE * SBMP * SBRP * 2);  // 41 MB
    __bf16* wdistT    = (__bf16*)carve((size_t)HID * NBF * 2);
    __bf16* wprojT    = (__bf16*)carve((size_t)HID * F2 * 2);
    __bf16* wedgeT    = (__bf16*)carve((size_t)HID * HID * 2);
    __bf16* woutT     = (__bf16*)carve((size_t)F1 * HID * 2);
    float*  gate      = (float*) carve((size_t)EEDGE * HID * 4);
    __bf16* rot       = (__bf16*)carve((size_t)EY * F2 * 2);              // 195 MB
    __bf16* msg1      = (__bf16*)carve((size_t)EY * HID * 2);
    __bf16* msg2      = (__bf16*)carve((size_t)EY * HID * 2);
    __bf16* msg3      = (__bf16*)carve((size_t)EY * F1 * 2);              // 97 MB
    __bf16* meanT     = (__bf16*)carve((size_t)EEDGE * CCH * SBRP * 2);   // 82 MB

    auto blocks_for = [](long n, int cap) {
        long b = (n + 255) / 256;
        return (int)(b > cap ? cap : b);
    };

    // 1) Layout-aware bf16 conversions
    {
        long n;
        n = (long)N_NODE * CCH * SBP;
        cvt_x_tpad<<<blocks_for(n, 8192), 256, 0, stream>>>(x, xT);
        n = (long)EEDGE * NBF;
        cvt_f32_bf16<<<blocks_for(n, 4096), 256, 0, stream>>>(x_edge, xe_bf, n);
        n = (long)EY * SBRP * SBP;
        cvt_pad<<<blocks_for(n, 8192), 256, 0, stream>>>(wigner, wig_pad, EY, SBR, SB, SBRP, SBP);
        n = (long)EEDGE * SBMP * SBRP;
        cvt_pad<<<blocks_for(n, 8192), 256, 0, stream>>>(wiginv, wiginv_p, EEDGE, SB, SBR, SBMP, SBRP);
        cvt_transpose<<<blocks_for((long)NBF * HID, 2048), 256, 0, stream>>>(w_dist, wdistT, NBF, HID);
        cvt_transpose<<<blocks_for((long)F2 * HID, 4096), 256, 0, stream>>>(w_proj, wprojT, F2, HID);
        cvt_transpose<<<blocks_for((long)HID * HID, 1024), 256, 0, stream>>>(w_edge, wedgeT, HID, HID);
        cvt_transpose<<<blocks_for((long)HID * F1, 2048), 256, 0, stream>>>(w_out, woutT, HID, F1);
    }

    // 2) distance gate: silu(x_edge @ w_dist + b_dist) -> f32 [E, H]
    {
        dim3 g((EEDGE + 127) / 128, HID / 128);
        gemm_bf16_silu<<<g, 256, 0, stream>>>(xe_bf, wdistT, b_dist, nullptr,
                                              gate, EEDGE, HID, NBF, 0);
    }

    // 3) rotate both endpoints with per-(edge,rot) Wigner matrices
    {
        dim3 g(EY, 2);
        rotate_wmma<<<g, 64, 0, stream>>>(wig_pad, xT, eidx, rot);
    }

    // 4) proj GEMM, fused bias+silu+gate: [EY, F2] @ [F2, H] -> bf16 [EY, H]
    {
        dim3 g((EY + 127) / 128, HID / 128);
        gemm_bf16_silu<<<g, 256, 0, stream>>>(rot, wprojT, b_proj, gate,
                                              msg1, EY, HID, F2, 1);
    }

    // 5) edge GEMM: silu(msg1 @ w_edge + b_edge) -> bf16 [EY, H]
    {
        dim3 g((EY + 127) / 128, HID / 128);
        gemm_bf16_silu<<<g, 256, 0, stream>>>(msg1, wedgeT, b_edge, nullptr,
                                              msg2, EY, HID, HID, 2);
    }

    // 6) out GEMM: silu(msg2 @ w_out + b_out) -> bf16 [EY, F1]
    {
        dim3 g((EY + 127) / 128, F1 / 128);
        gemm_bf16_silu<<<g, 256, 0, stream>>>(msg2, woutT, b_out, nullptr,
                                              msg3, EY, F1, HID, 2);
    }

    // 7) mean over Y rotated copies -> bf16 transposed+padded [E][128][32]
    {
        long n = (long)EEDGE * CCH * SBRP;
        mean_y_tpad<<<blocks_for(n, 8192), 256, 0, stream>>>(msg3, meanT);
    }

    // 8) inverse rotation -> f32 output [E, SB, C]
    rotate_inv_wmma<<<EEDGE, 128, 0, stream>>>(wiginv_p, meanT, out);
}
